// Attention_63359357550974
// MI455X (gfx1250) — matrix operand order repaired and checked by reading, not compiled, not run on hardware
//
#include <hip/hip_runtime.h>
#include <hip/hip_bf16.h>

typedef __attribute__((ext_vector_type(16))) _Float16 v16h;
typedef __attribute__((ext_vector_type(8)))  _Float16 v8h;
typedef __attribute__((ext_vector_type(4)))  _Float16 v4h;
typedef __attribute__((ext_vector_type(8)))  float    v8f;
typedef __attribute__((ext_vector_type(4)))  float    f4;
typedef __attribute__((ext_vector_type(4)))  unsigned int u32x4;
typedef __attribute__((ext_vector_type(8)))  int      i32x8;
typedef __attribute__((ext_vector_type(4)))  int      i32x4;

#define DIMD 2048
#define NQKV 6144
#define LSEQ 1024
#define KVSEQ 2048
#define NH 16
#define DH 128

#if defined(__has_builtin)
#if __has_builtin(__builtin_amdgcn_tensor_load_to_lds) && \
    __has_builtin(__builtin_amdgcn_s_wait_tensorcnt)
#define USE_TDM 1
#endif
#endif

__device__ __forceinline__ v8f wmma16(v16h a, v16h b, v8f c) {
    return __builtin_amdgcn_wmma_f32_16x16x32_f16(false, a, false, b, (short)0, c, false, false);
}

// 16 contiguous halves -> fragment (two aligned v8h loads)
__device__ __forceinline__ v16h ld16(const _Float16* p) {
    v8h lo = *(const v8h*)p;
    v8h hi = *(const v8h*)(p + 8);
    v16h r;
#pragma unroll
    for (int i = 0; i < 8; ++i) { r[i] = lo[i]; r[i + 8] = hi[i]; }
    return r;
}

// two separate 8-half runs -> fragment (A callers pass adjacent runs: contiguous k = 16h + s)
__device__ __forceinline__ v16h ld2x8(const _Float16* p0, const _Float16* p1) {
    v8h lo = *(const v8h*)p0;
    v8h hi = *(const v8h*)p1;
    v16h r;
#pragma unroll
    for (int i = 0; i < 8; ++i) { r[i] = lo[i]; r[i + 8] = hi[i]; }
    return r;
}

#ifdef USE_TDM
// TDM: DMA a 64-row x 32-half f16 tile (row stride 2048 halves in memory) into
// LDS with hardware padding of 4 dwords after every 16 dwords -> LDS row
// stride 40 halves (bank-conflict-free fragment reads).
__device__ __forceinline__ void tdm_load_a_tile(const _Float16* gbase, unsigned lds_off) {
    unsigned long long ga = (unsigned long long)(size_t)gbase;
    u32x4 g0;
    g0[0] = 1u;                                  // count=1, user descriptor
    g0[1] = lds_off;                             // lds_addr (bytes)
    g0[2] = (unsigned)(ga & 0xffffffffu);        // global_addr[31:0]
    g0[3] = (unsigned)((ga >> 32) & 0x1ffffffu)  // global_addr[56:32]
          | (2u << 30);                          // type = 2 ("image")
    i32x8 g1;
    g1[0] = (1 << 16)        // data_size = 2 bytes
          | (1 << 20)        // pad_enable
          | (3 << 22)        // pad_interval: 16 dwords (one 32-half row)
          | (3 << 25);       // pad_amount: 4 dwords (8 halves)
    g1[1] = (int)(2048u << 16);        // tensor_dim0[15:0] = 2048
    g1[2] = (int)(2048u << 16);        // tensor_dim0[31:16]=0 | tensor_dim1[15:0]=2048
    g1[3] = (int)(32u << 16);          // tensor_dim1[31:16]=0 | tile_dim0 = 32
    g1[4] = 64;                        // tile_dim1 = 64 | tile_dim2 = 0
    g1[5] = 2048;                      // tensor_dim0_stride[31:0]
    g1[6] = 0;                         // stride0[47:32] | tensor_dim1_stride[15:0]
    g1[7] = 0;
    i32x4 g2 = {0, 0, 0, 0};
    i32x4 g3 = {0, 0, 0, 0};
    i32x8 g4 = {0, 0, 0, 0, 0, 0, 0, 0};
    // 6-arg form (clang-23 / therock headers): (g0, g1, g2, g3, extra, cpol)
    __builtin_amdgcn_tensor_load_to_lds(g0, g1, g2, g3, g4, 0);
}
#endif

// ---------------------------------------------------------------------------
// Kernel 1: copy active caches into rows [0,1024) of output K/V (b,h,2048,128)
// ---------------------------------------------------------------------------
__global__ __launch_bounds__(256) void copy_caches(const float* __restrict__ kac,
                                                   const float* __restrict__ vac,
                                                   float* __restrict__ outK,
                                                   float* __restrict__ outV) {
    const f4* ks = (const f4*)kac;
    const f4* vs = (const f4*)vac;
    f4* kd = (f4*)outK;
    f4* vd = (f4*)outV;
    const size_t total = (size_t)2 * NH * LSEQ * DH / 4;  // 1,048,576 float4 per tensor
    for (size_t i = (size_t)blockIdx.x * blockDim.x + threadIdx.x; i < total;
         i += (size_t)gridDim.x * blockDim.x) {
        size_t bh  = i >> 15;
        size_t rem = i & 32767;
        kd[bh * 65536 + rem] = ks[i];
        vd[bh * 65536 + rem] = vs[i];
    }
}

// ---------------------------------------------------------------------------
// Kernel 2: QKV GEMM.  C(2048x6144) = X(2048x2048) @ Wqkv + bqkv
// ---------------------------------------------------------------------------
__global__ __launch_bounds__(256) void qkv_gemm(const float* __restrict__ X,
                                                const float* __restrict__ W,
                                                const float* __restrict__ bias,
                                                float* __restrict__ outK,
                                                float* __restrict__ outV,
                                                _Float16* __restrict__ qh) {
    __shared__ _Float16 As[64 * 40];    // A tile 64x32, padded stride 40
    __shared__ _Float16 Bs[128 * 40];   // B tile transposed: [n][k], stride 40
    const int tid  = threadIdx.x;
    const int lane = tid & 31;
    const int w    = tid >> 5;
    const int wm   = w >> 2;
    const int wn   = w & 3;
    const int m0   = blockIdx.y * 64;
    const int n0   = blockIdx.x * 128;
    const int nn   = lane & 15;
    const int mb   = (lane >> 4) * 8;
    const int kbA  = (lane >> 4) * 16;
    const int kbB  = (lane >> 4) * 16;

    v8f acc[2][2] = {};

    for (int k0 = 0; k0 < DIMD; k0 += 32) {
        {
            int r = tid >> 3;
            int c = (tid & 7) * 4;
#pragma unroll
            for (int rr = 0; rr < 2; ++rr) {
                int row = r + rr * 32;
                f4 v = *(const f4*)(X + (size_t)(m0 + row) * DIMD + k0 + c);
                v4h p;
                p[0] = (_Float16)v.x; p[1] = (_Float16)v.y;
                p[2] = (_Float16)v.z; p[3] = (_Float16)v.w;
                *(v4h*)&As[row * 40 + c] = p;
            }
        }
        {
            int k  = tid >> 3;
            int nb = (tid & 7) * 16;
            const float* src = W + (size_t)(k0 + k) * NQKV + n0 + nb;
#pragma unroll
            for (int i = 0; i < 4; ++i) {
                f4 v = *(const f4*)(src + i * 4);
                Bs[(nb + i * 4 + 0) * 40 + k] = (_Float16)v.x;
                Bs[(nb + i * 4 + 1) * 40 + k] = (_Float16)v.y;
                Bs[(nb + i * 4 + 2) * 40 + k] = (_Float16)v.z;
                Bs[(nb + i * 4 + 3) * 40 + k] = (_Float16)v.w;
            }
        }
        __syncthreads();

        v16h af[2], bf[2];
#pragma unroll
        for (int i = 0; i < 2; ++i) {
            const _Float16* pa = &As[(wm * 32 + i * 16 + nn) * 40];
            af[i] = ld2x8(pa + kbA, pa + kbA + 8);
        }
#pragma unroll
        for (int j = 0; j < 2; ++j)
            bf[j] = ld16(&Bs[(wn * 32 + j * 16 + nn) * 40 + kbB]);
#pragma unroll
        for (int i = 0; i < 2; ++i)
#pragma unroll
            for (int j = 0; j < 2; ++j)
                acc[i][j] = wmma16(af[i], bf[j], acc[i][j]);
        __syncthreads();
    }

    const float scale = 0.08838834764831845f;  // 1/sqrt(128)
#pragma unroll
    for (int i = 0; i < 2; ++i) {
#pragma unroll
        for (int j = 0; j < 2; ++j) {
            int col  = n0 + wn * 32 + j * 16 + nn;
            int sec  = col >> 11;
            int h    = (col >> 7) & 15;
            int dd   = col & 127;
            float bv = bias[col];
            int rbase = m0 + wm * 32 + i * 16 + mb;
#pragma unroll
            for (int jj = 0; jj < 8; ++jj) {
                int r  = rbase + jj;
                int bb = r >> 10, ll = r & 1023;
                float val = acc[i][j][jj] + bv;
                if (sec == 0) {
                    qh[(((size_t)(bb * NH + h)) * LSEQ + ll) * DH + dd] =
                        (_Float16)(val * scale);
                } else if (sec == 1) {
                    outK[(((size_t)(bb * NH + h)) * KVSEQ + LSEQ + ll) * DH + dd] = val;
                } else {
                    outV[(((size_t)(bb * NH + h)) * KVSEQ + LSEQ + ll) * DH + dd] = val;
                }
            }
        }
    }
}

// ---------------------------------------------------------------------------
// Kernel 3: flash attention.  block = (bh, 128 q rows), 8 waves x 16 q rows.
// ---------------------------------------------------------------------------
__global__ __launch_bounds__(256) void attn(const _Float16* __restrict__ qh,
                                            const float* __restrict__ Kc,
                                            const float* __restrict__ Vc,
                                            _Float16* __restrict__ oh) {
    __shared__ _Float16 Kt[32 * 136];
    __shared__ _Float16 Vt[128 * 40];
    __shared__ _Float16 Pw[8][16 * 40];

    const int tid  = threadIdx.x;
    const int lane = tid & 31;
    const int w    = tid >> 5;
    const int bh   = blockIdx.x;
    const int qt   = blockIdx.y;
    const int nn   = lane & 15;
    const int mb   = (lane >> 4) * 8;
    const int kbA  = (lane >> 4) * 16;
    const int kbB  = (lane >> 4) * 16;
    const int qrow0 = qt * 128 + w * 16;

    const float* Kb = Kc + (size_t)bh * KVSEQ * DH;
    const float* Vb = Vc + (size_t)bh * KVSEQ * DH;
    const _Float16* Qb = qh + ((size_t)bh * LSEQ + qrow0) * DH;

    v16h qa[4];
    {
        const _Float16* qr = Qb + nn * DH;
#pragma unroll
        for (int f = 0; f < 4; ++f)
            qa[f] = ld2x8(qr + f * 32 + kbA, qr + f * 32 + kbA + 8);
    }

    v8f o[8] = {};
    float mrow[8], lrow[8];
#pragma unroll
    for (int j = 0; j < 8; ++j) { mrow[j] = -3.0e38f; lrow[j] = 0.0f; }

    for (int kc = 0; kc < KVSEQ / 32; ++kc) {
        const int kcol0 = kc * 32;
        {
            int r  = tid >> 3;
            int d0 = (tid & 7) * 16;
            const float* ksrc = Kb + (size_t)(kcol0 + r) * DH + d0;
            const float* vsrc = Vb + (size_t)(kcol0 + r) * DH + d0;
#pragma unroll
            for (int i = 0; i < 4; ++i) {
                f4 kv = *(const f4*)(ksrc + i * 4);
                v4h p;
                p[0] = (_Float16)kv.x; p[1] = (_Float16)kv.y;
                p[2] = (_Float16)kv.z; p[3] = (_Float16)kv.w;
                *(v4h*)&Kt[r * 136 + d0 + i * 4] = p;
                f4 vv = *(const f4*)(vsrc + i * 4);
                Vt[(d0 + i * 4 + 0) * 40 + r] = (_Float16)vv.x;
                Vt[(d0 + i * 4 + 1) * 40 + r] = (_Float16)vv.y;
                Vt[(d0 + i * 4 + 2) * 40 + r] = (_Float16)vv.z;
                Vt[(d0 + i * 4 + 3) * 40 + r] = (_Float16)vv.w;
            }
        }
        __syncthreads();

        v8f s0 = {}, s1 = {};
#pragma unroll
        for (int s = 0; s < 4; ++s) {
            v16h b0 = ld16(&Kt[(nn) * 136 + s * 32 + kbB]);
            v16h b1 = ld16(&Kt[(16 + nn) * 136 + s * 32 + kbB]);
            s0 = wmma16(qa[s], b0, s0);
            s1 = wmma16(qa[s], b1, s1);
        }

#pragma unroll
        for (int j = 0; j < 8; ++j) {
            float mx = fmaxf(s0[j], s1[j]);
            mx = fmaxf(mx, __shfl_xor(mx, 1, 32));
            mx = fmaxf(mx, __shfl_xor(mx, 2, 32));
            mx = fmaxf(mx, __shfl_xor(mx, 4, 32));
            mx = fmaxf(mx, __shfl_xor(mx, 8, 32));
            float mnew  = fmaxf(mrow[j], mx);
            float alpha = __expf(mrow[j] - mnew);
            float p0 = __expf(s0[j] - mnew);
            float p1 = __expf(s1[j] - mnew);
            s0[j] = p0; s1[j] = p1;
            float rs = p0 + p1;
            rs += __shfl_xor(rs, 1, 32);
            rs += __shfl_xor(rs, 2, 32);
            rs += __shfl_xor(rs, 4, 32);
            rs += __shfl_xor(rs, 8, 32);
            lrow[j] = lrow[j] * alpha + rs;
            mrow[j] = mnew;
#pragma unroll
            for (int c = 0; c < 8; ++c) o[c][j] *= alpha;
        }

#pragma unroll
        for (int j = 0; j < 8; ++j) {
            Pw[w][(mb + j) * 40 + nn]      = (_Float16)s0[j];
            Pw[w][(mb + j) * 40 + 16 + nn] = (_Float16)s1[j];
        }
        v16h pa = ld2x8(&Pw[w][nn * 40 + kbA], &Pw[w][nn * 40 + kbA + 8]);

#pragma unroll
        for (int c = 0; c < 8; ++c) {
            v16h bv = ld16(&Vt[(c * 16 + nn) * 40 + kbB]);
            o[c] = wmma16(pa, bv, o[c]);
        }
        __syncthreads();
    }

    const int b = bh >> 4, h = bh & 15;
#pragma unroll
    for (int j = 0; j < 8; ++j) {
        float inv = 1.0f / lrow[j];
        size_t row = (size_t)b * LSEQ + qrow0 + mb + j;
        _Float16* dst = oh + (row * NH + h) * DH + nn;
#pragma unroll
        for (int c = 0; c < 8; ++c) dst[c * 16] = (_Float16)(o[c][j] * inv);
    }
}

// ---------------------------------------------------------------------------
// Kernel 4: output GEMM. out = o_f16(2048x2048) @ Wout + bout (f32 out).
// A tile staged by the Tensor Data Mover (double-buffered) when available.
// ---------------------------------------------------------------------------
__global__ __launch_bounds__(256) void out_gemm(const _Float16* __restrict__ A,
                                                const float* __restrict__ W,
                                                const float* __restrict__ bias,
                                                float* __restrict__ out) {
    __shared__ _Float16 As[2][64 * 40];
    __shared__ _Float16 Bs[128 * 40];
    const int tid  = threadIdx.x;
    const int lane = tid & 31;
    const int w    = tid >> 5;
    const int wm   = w >> 2;
    const int wn   = w & 3;
    const int m0   = blockIdx.y * 64;
    const int n0   = blockIdx.x * 128;
    const int nn   = lane & 15;
    const int mb   = (lane >> 4) * 8;
    const int kbA  = (lane >> 4) * 16;
    const int kbB  = (lane >> 4) * 16;

    v8f acc[2][2] = {};
    int cur = 0;

#ifdef USE_TDM
    if (w == 0)
        tdm_load_a_tile(A + (size_t)m0 * DIMD, (unsigned)(size_t)&As[0][0]);
#endif

    for (int k0 = 0; k0 < DIMD; k0 += 32) {
        // stage B transposed (f32 -> f16)
        {
            int k  = tid >> 3;
            int nb = (tid & 7) * 16;
            const float* src = W + (size_t)(k0 + k) * DIMD + n0 + nb;
#pragma unroll
            for (int i = 0; i < 4; ++i) {
                f4 v = *(const f4*)(src + i * 4);
                Bs[(nb + i * 4 + 0) * 40 + k] = (_Float16)v.x;
                Bs[(nb + i * 4 + 1) * 40 + k] = (_Float16)v.y;
                Bs[(nb + i * 4 + 2) * 40 + k] = (_Float16)v.z;
                Bs[(nb + i * 4 + 3) * 40 + k] = (_Float16)v.w;
            }
        }
#ifdef USE_TDM
        if (w == 0) __builtin_amdgcn_s_wait_tensorcnt(0);
#else
        {
            int row = tid >> 2;
            int c8  = (tid & 3) * 8;
            *(v8h*)&As[0][row * 40 + c8] =
                *(const v8h*)(A + (size_t)(m0 + row) * DIMD + k0 + c8);
        }
#endif
        __syncthreads();
#ifdef USE_TDM
        if (w == 0 && k0 + 32 < DIMD)
            tdm_load_a_tile(A + (size_t)m0 * DIMD + k0 + 32,
                            (unsigned)(size_t)&As[cur ^ 1][0]);
#endif

        v16h af[2], bf[2];
#pragma unroll
        for (int i = 0; i < 2; ++i) {
            const _Float16* pa = &As[cur][(wm * 32 + i * 16 + nn) * 40];
            af[i] = ld2x8(pa + kbA, pa + kbA + 8);
        }
#pragma unroll
        for (int j = 0; j < 2; ++j)
            bf[j] = ld16(&Bs[(wn * 32 + j * 16 + nn) * 40 + kbB]);
#pragma unroll
        for (int i = 0; i < 2; ++i)
#pragma unroll
            for (int j = 0; j < 2; ++j)
                acc[i][j] = wmma16(af[i], bf[j], acc[i][j]);
        __syncthreads();
#ifdef USE_TDM
        cur ^= 1;
#endif
    }

#pragma unroll
    for (int i = 0; i < 2; ++i) {
#pragma unroll
        for (int j = 0; j < 2; ++j) {
            int col  = n0 + wn * 32 + j * 16 + nn;
            float bv = bias[col];
            int rbase = m0 + wm * 32 + i * 16 + mb;
#pragma unroll
            for (int jj = 0; jj < 8; ++jj)
                out[(size_t)(rbase + jj) * DIMD + col] = acc[i][j][jj] + bv;
        }
    }
}

// ---------------------------------------------------------------------------
extern "C" void kernel_launch(void* const* d_in, const int* in_sizes, int n_in,
                              void* d_out, int out_size, void* d_ws, size_t ws_size,
                              hipStream_t stream) {
    const float* x    = (const float*)d_in[0];
    const float* kac  = (const float*)d_in[1];
    const float* vac  = (const float*)d_in[2];
    const float* Wqkv = (const float*)d_in[3];
    const float* bqkv = (const float*)d_in[4];
    const float* Wout = (const float*)d_in[5];
    const float* bout = (const float*)d_in[6];

    float* out  = (float*)d_out;
    float* outK = out + (size_t)2 * LSEQ * DIMD;
    float* outV = outK + (size_t)2 * NH * KVSEQ * DH;

    _Float16* qh = (_Float16*)d_ws;
    _Float16* oh = qh + (size_t)2 * NH * LSEQ * DH;

    copy_caches<<<2048, 256, 0, stream>>>(kac, vac, outK, outV);
    qkv_gemm<<<dim3(NQKV / 128, DIMD / 64), 256, 0, stream>>>(x, Wqkv, bqkv,
                                                              outK, outV, qh);
    attn<<<dim3(2 * NH, LSEQ / 128), 256, 0, stream>>>(qh, outK, outV, oh);
    out_gemm<<<dim3(DIMD / 128, DIMD / 64), 256, 0, stream>>>(oh, Wout, bout, out);
}
